// MP_JEPA_76957224010257
// MI455X (gfx1250) — compile-verified, matrix operand order checked by reading, not run on hardware
//
#include <hip/hip_runtime.h>
#include <hip/hip_bf16.h>

// ---------- problem constants ----------
#define NN   100000
#define DD   128
#define ZZ   128
#define TT   1024
#define KK   256
#define HID  512
#define DIN  384   // D + 2Z

// ---------- WMMA types ----------
typedef __attribute__((ext_vector_type(16))) __bf16 v16bf;
typedef __attribute__((ext_vector_type(8)))  float  v8f;

union Frag { uint4 u[2]; v16bf v; };

__device__ __forceinline__ unsigned short f2bf(float f) {
    union { float f; unsigned u; } x; x.f = f;
    unsigned r = x.u + 0x7FFFu + ((x.u >> 16) & 1u);   // round-to-nearest-even
    return (unsigned short)(r >> 16);
}

// ---------- LDS geometry (bf16 elements) ----------
#define ASTRIDE 408                    // 64 rows x 816B  (51 uint4/row; 204 dwords -> conflict-friendly)
#define HSTRIDE 520                    // 64 rows x 1040B (260 dwords)
#define A_BYTES (64 * ASTRIDE * 2)     // 52224
#define H_BYTES (64 * HSTRIDE * 2)     // 66560
#define RED_BYTES (256 * 4)
#define SMEM_BYTES (A_BYTES + H_BYTES + RED_BYTES)

// ===================== prep kernels =====================

// posz[n][z] = bf16( pe[n,:4] @ W_pe[:,z] + zparam[z] ),  one block per node
__global__ void k_posz(const float* __restrict__ pe, const float* __restrict__ wpe,
                       const float* __restrict__ zp, unsigned short* __restrict__ posz) {
    int n = blockIdx.x;
    int zi = threadIdx.x;            // 0..127
    float4 p = ((const float4*)pe)[n];
    float v = p.x * wpe[0 * ZZ + zi] + p.y * wpe[1 * ZZ + zi] +
              p.z * wpe[2 * ZZ + zi] + p.w * wpe[3 * ZZ + zi] + zp[zi];
    posz[(size_t)n * ZZ + zi] = f2bf(v);
}

// plain f32 -> bf16 convert
__global__ void k_cvt(const float* __restrict__ src, unsigned short* __restrict__ dst, int n) {
    int i = blockIdx.x * blockDim.x + threadIdx.x;
    if (i < n) dst[i] = f2bf(src[i]);
}

// W1 [384][512] -> W1T bf16 [512][384]
__global__ void k_w1t(const float* __restrict__ w1, unsigned short* __restrict__ w1t) {
    int i = blockIdx.x * blockDim.x + threadIdx.x;   // over 512*384
    int n = i / DIN, k = i % DIN;
    w1t[i] = f2bf(w1[(size_t)k * HID + n]);
}

// W2 [512][128] -> W2T bf16 [128][512]
__global__ void k_w2t(const float* __restrict__ w2, unsigned short* __restrict__ w2t) {
    int i = blockIdx.x * blockDim.x + threadIdx.x;   // over 128*512
    int n = i / HID, k = i % HID;
    w2t[i] = f2bf(w2[(size_t)k * DD + n]);
}

// ===================== main kernel =====================
// 1 block per target; 256 threads = 8 wave32.
__global__ __launch_bounds__(256)
void k_main(const unsigned short* __restrict__ ctxb,   // [N][128] bf16
            const unsigned short* __restrict__ poszb,  // [N][128] bf16
            const unsigned short* __restrict__ w1t,    // [512][384] bf16
            const unsigned short* __restrict__ w2t,    // [128][512] bf16
            const float* __restrict__ b1, const float* __restrict__ b2,
            const float* __restrict__ tgt_emb,         // [N][128] f32
            const int* __restrict__ sub_nodes, const int* __restrict__ sub_counts,
            const int* __restrict__ target_nodes, float* __restrict__ out) {
    extern __shared__ char smem[];
    unsigned short* Alds = (unsigned short*)smem;                 // [64][ASTRIDE]
    unsigned short* Hlds = (unsigned short*)(smem + A_BYTES);     // [64][HSTRIDE]
    float*          red  = (float*)(smem + A_BYTES + H_BYTES);

    const int t    = blockIdx.x;
    const int tid  = threadIdx.x;
    const int lane = tid & 31;
    const int wave = tid >> 5;
    const int l16  = lane & 15;
    const int half = lane >> 4;      // selects K sub-chunk per ISA A/B layout

    const int tnode = target_nodes[t];
    int cnt = sub_counts[t]; if (cnt < 1) cnt = 1;

    float lsum = 0.f;

    for (int c = 0; c < 4; ++c) {
        // ---- stage A chunk: rows c*64 .. c*64+63, 384 bf16 each ----
        {
            int rowi = tid >> 2;             // 0..63
            int quad = tid & 3;              // 4 threads per row, 12 uint4 each
            int node = sub_nodes[t * KK + c * 64 + rowi];
            const uint4* ctx4 = (const uint4*)ctxb  + (size_t)node  * 16;
            const uint4* pn4  = (const uint4*)poszb + (size_t)node  * 16;
            const uint4* pt4  = (const uint4*)poszb + (size_t)tnode * 16;
            uint4* dst = (uint4*)(Alds + rowi * ASTRIDE);
#pragma unroll
            for (int j = 0; j < 12; ++j) {
                int idx = quad * 12 + j;     // 0..47
                uint4 v;
                if (idx < 16)      v = ctx4[idx];
                else if (idx < 32) v = pn4[idx - 16];
                else               v = pt4[idx - 32];
                dst[idx] = v;
            }
        }
        __syncthreads();

        // ---- GEMM1: H[64][512] = relu(A[64][384] @ W1 + b1), bf16 out to LDS ----
        // wave -> ngroup = wave&3 (128 cols = 8 N-tiles), mgroup = wave>>2 (32 rows),
        // done as two passes of 1 M-tile x 8 N-tiles: one A fragment feeds 8 WMMAs
        // -> 128B of LDS traffic per WMMA.  W1T re-reads hit L2.
        {
            const int ngroup = wave & 3;           // cols ngroup*128 .. +127
            const int mgroup = wave >> 2;          // rows mgroup*32 .. +31
            const unsigned short* bbase = w1t + (size_t)(ngroup * 128 + l16) * DIN;

#pragma unroll 1
            for (int p = 0; p < 2; ++p) {          // M-tile: rows mgroup*32 + p*16 ..
                const int mrow = mgroup * 32 + p * 16;
                v8f acc[8];
#pragma unroll
                for (int i = 0; i < 8; ++i) acc[i] = (v8f){0,0,0,0,0,0,0,0};

#pragma unroll 1
                for (int ks = 0; ks < 12; ++ks) {
                    const int k0 = ks * 32;
                    const int o0 = (k0 + half * 8) >> 3, o1 = (k0 + 16 + half * 8) >> 3;

                    Frag a;
                    const uint4* r = (const uint4*)(Alds + (mrow + l16) * ASTRIDE);
                    a.u[0] = r[o0]; a.u[1] = r[o1];

                    Frag b[8];
#pragma unroll
                    for (int nt = 0; nt < 8; ++nt) {
                        const uint4* bp = (const uint4*)(bbase + (size_t)nt * 16 * DIN);
                        b[nt].u[0] = bp[o0]; b[nt].u[1] = bp[o1];
                    }
#pragma unroll
                    for (int nt = 0; nt < 8; ++nt)
                        acc[nt] = __builtin_amdgcn_wmma_f32_16x16x32_bf16(
                            false, a.v, false, b[nt].v, (short)0, acc[nt], false, false);
                }
                // epilogue: bias + relu + bf16 store to Hlds
#pragma unroll
                for (int nt = 0; nt < 8; ++nt) {
                    const int ncol = ngroup * 128 + nt * 16 + l16;
                    const float bv = b1[ncol];
#pragma unroll
                    for (int g = 0; g < 8; ++g) {
                        float h = acc[nt][g] + bv;
                        h = h > 0.f ? h : 0.f;
                        int row = mrow + g + half * 8;       // C/D: vgpr g -> M=g(+8)
                        Hlds[row * HSTRIDE + ncol] = f2bf(h);
                    }
                }
            }
        }
        __syncthreads();

        // ---- GEMM2: P[64][128] = H @ W2 + b2 ; loss accumulate ----
        // wave -> mtile = wave&3 (16 rows), ngroup = wave>>2 (64 cols = 4 N-tiles):
        // one A fragment feeds 4 WMMAs -> 256B LDS per WMMA.
        {
            const int mtile  = wave & 3;           // rows mtile*16 .. +15
            const int ngroup = wave >> 2;          // cols ngroup*64 .. +63
            const unsigned short* bbase = w2t + (size_t)(ngroup * 64 + l16) * HID;

            v8f acc[4];
#pragma unroll
            for (int i = 0; i < 4; ++i) acc[i] = (v8f){0,0,0,0,0,0,0,0};

#pragma unroll 1
            for (int ks = 0; ks < 16; ++ks) {
                const int k0 = ks * 32;
                const int o0 = (k0 + half * 8) >> 3, o1 = (k0 + 16 + half * 8) >> 3;

                Frag a;
                const uint4* r = (const uint4*)(Hlds + (mtile * 16 + l16) * HSTRIDE);
                a.u[0] = r[o0]; a.u[1] = r[o1];

                Frag b[4];
#pragma unroll
                for (int nt = 0; nt < 4; ++nt) {
                    const uint4* bp = (const uint4*)(bbase + (size_t)nt * 16 * HID);
                    b[nt].u[0] = bp[o0]; b[nt].u[1] = bp[o1];
                }
#pragma unroll
                for (int nt = 0; nt < 4; ++nt)
                    acc[nt] = __builtin_amdgcn_wmma_f32_16x16x32_bf16(
                        false, a.v, false, b[nt].v, (short)0, acc[nt], false, false);
            }
#pragma unroll
            for (int nt = 0; nt < 4; ++nt) {
                const int ncol = ngroup * 64 + nt * 16 + l16;
                const float bv = b2[ncol];
                const float tv = tgt_emb[(size_t)tnode * DD + ncol];
#pragma unroll
                for (int g = 0; g < 8; ++g) {
                    int row = c * 64 + mtile * 16 + g + half * 8;
                    if (row < cnt) {
                        float d = acc[nt][g] + bv - tv;
                        lsum += d * d;
                    }
                }
            }
        }
        __syncthreads();
    }

    // ---- block reduction + scaled atomic ----
    red[tid] = lsum;
    __syncthreads();
    for (int s = 128; s > 0; s >>= 1) {
        if (tid < s) red[tid] += red[tid + s];
        __syncthreads();
    }
    if (tid == 0) atomicAdd(out, red[0] / ((float)cnt * (float)DD));
}

// ===================== launch =====================
extern "C" void kernel_launch(void* const* d_in, const int* in_sizes, int n_in,
                              void* d_out, int out_size, void* d_ws, size_t ws_size,
                              hipStream_t stream) {
    const float* pe      = (const float*)d_in[0];
    const float* ctx     = (const float*)d_in[1];
    const float* tgt_emb = (const float*)d_in[2];
    const int*   sub_nodes    = (const int*)d_in[3];
    const int*   sub_counts   = (const int*)d_in[4];
    const int*   target_nodes = (const int*)d_in[5];
    const float* zp   = (const float*)d_in[6];
    const float* wpe  = (const float*)d_in[7];
    const float* w1   = (const float*)d_in[8];
    const float* b1   = (const float*)d_in[9];
    const float* w2   = (const float*)d_in[10];
    const float* b2   = (const float*)d_in[11];
    float* out = (float*)d_out;

    // workspace layout (bf16 tables)
    char* ws = (char*)d_ws;
    unsigned short* poszb = (unsigned short*)(ws);                                   // N*Z*2   = 25.6MB
    unsigned short* ctxb  = (unsigned short*)(ws + (size_t)NN * ZZ * 2);             // N*D*2   = 25.6MB
    unsigned short* w1t   = (unsigned short*)(ws + (size_t)NN * (ZZ + DD) * 2);      // 512*384*2
    unsigned short* w2t   = (unsigned short*)(ws + (size_t)NN * (ZZ + DD) * 2 + (size_t)HID * DIN * 2);

    hipMemsetAsync(d_out, 0, sizeof(float), stream);

    k_posz<<<NN, ZZ, 0, stream>>>(pe, wpe, zp, poszb);
    k_cvt<<<(NN * DD + 255) / 256, 256, 0, stream>>>(ctx, ctxb, NN * DD);
    k_w1t<<<(HID * DIN + 255) / 256, 256, 0, stream>>>(w1, w1t);
    k_w2t<<<(DD * HID + 255) / 256, 256, 0, stream>>>(w2, w2t);

    k_main<<<TT, 256, SMEM_BYTES, stream>>>(ctxb, poszb, w1t, w2t, b1, b2, tgt_emb,
                                            sub_nodes, sub_counts, target_nodes, out);
}
